// LanguageModel_43095701848102
// MI455X (gfx1250) — matrix operand order repaired and checked by reading, not compiled, not run on hardware
//
#include <hip/hip_runtime.h>

// ---------------- problem constants (from reference) ----------------
#define C_   384
#define H_   6
#define HD_  64
#define L_   6
#define V_   65
#define VP_  80              // V padded to multiple of 16
#define B_   128
#define T_   256
#define FF_  1536
#define BT_  (B_ * T_)       // 32768
#define SCALE_ 0.05103103630798287f   // 384^-0.5
#define EPS_ 1e-5f

typedef __attribute__((ext_vector_type(16))) _Float16 v16h;
typedef __attribute__((ext_vector_type(8)))  _Float16 v8h;
typedef __attribute__((ext_vector_type(8)))  float    v8f;

__device__ __forceinline__ v8f wmma_f16(v16h a, v16h b, v8f c) {
  return __builtin_amdgcn_wmma_f32_16x16x32_f16(false, a, false, b, (short)0, c,
                                                false, false);
}

// Scalar (SGPR) wave index: tile/row branches become s_cbranch, not exec masks.
__device__ __forceinline__ int wave_id() {
  return __builtin_amdgcn_readfirstlane((int)(threadIdx.x >> 5));
}

// A fragment 16x32 f16, row-major A[lda]: single contiguous 32B load.
// a[i] <-> K = 16*hi + i; m = lane&15.
__device__ __forceinline__ v16h a_frag(const _Float16* __restrict__ A, int lda,
                                       int row0, int k0, unsigned lane) {
  int m = lane & 15, hi = lane >> 4;
  const _Float16* p = A + (size_t)(row0 + m) * lda + k0 + 16 * hi;
  return *(const v16h*)(p);
}

// B fragment 32x16 f16 from pre-transposed Bt[N, ldk]: single contiguous 32B load.
// b[i] <-> K = 16*hi + i of column n.
__device__ __forceinline__ v16h b_frag(const _Float16* __restrict__ Bt, int ldk,
                                       int n, int k0, int hi) {
  return *(const v16h*)(Bt + (size_t)n * ldk + k0 + 16 * hi);
}

// ---------------- generic GEMM: O[M,N] = A[M,K] @ Bt^T + bias (+relu) ---------
// A f16 [M,K]; Bt f16 [Npad,K] transposed+padded. 128 thr = 4 waves; each wave
// computes 2 M-tiles x 4 N-tiles (32x64): 8 WMMAs per k-step off 6 vector loads.
// FULLN=true: all 4 n-tiles valid (branch-free inner loop).
// OMODE: 0 fp32 row-major, 1 f16 row-major, 2 f16 transposed O[((r/T)*ldo+n)*T+r%T].
template <int OMODE, bool FULLN>
__global__ __launch_bounds__(128) void gemm_k(
    const _Float16* __restrict__ A, int lda,
    const _Float16* __restrict__ Bt,
    const float* __restrict__ bias,
    void* __restrict__ Ov, int ldo,
    int M, int N, int Npad, int K, int relu) {
  unsigned lane = threadIdx.x & 31;
  int wave = wave_id();
  int row0 = (blockIdx.y * 8 + wave * 2) * 16;   // 2 m-tiles per wave
  if (row0 >= M) return;
  int nb = blockIdx.x * 64;
  int hi = lane >> 4, nl = lane & 15;
  v8f acc0[4] = {}, acc1[4] = {};
  for (int k0 = 0; k0 < K; k0 += 32) {
    v16h a0 = a_frag(A, lda, row0, k0, lane);
    v16h a1 = a_frag(A, lda, row0 + 16, k0, lane);
#pragma unroll
    for (int t = 0; t < 4; ++t) {
      int n0 = nb + 16 * t;
      if (FULLN || n0 < Npad) {
        v16h b = b_frag(Bt, K, n0 + nl, k0, hi);
        acc0[t] = wmma_f16(a0, b, acc0[t]);
        acc1[t] = wmma_f16(a1, b, acc1[t]);
      }
    }
  }
#pragma unroll
  for (int mt = 0; mt < 2; ++mt) {
    int rb = row0 + 16 * mt;
#pragma unroll
    for (int t = 0; t < 4; ++t) {
      int n = nb + 16 * t + nl;
      if (n >= N) continue;
      float bv = bias ? bias[n] : 0.f;
      v8f* acc = mt ? acc1 : acc0;
#pragma unroll
      for (int j = 0; j < 8; ++j) {
        int r = rb + j + 8 * hi;
        float val = acc[t][j] + bv;
        if (relu) val = fmaxf(val, 0.f);
        if (OMODE == 0) {
          ((float*)Ov)[(size_t)r * ldo + n] = val;
        } else if (OMODE == 1) {
          ((_Float16*)Ov)[(size_t)r * ldo + n] = (_Float16)val;
        } else {
          int bb = r / T_, tt = r % T_;
          ((_Float16*)Ov)[((size_t)bb * ldo + n) * T_ + tt] = (_Float16)val;
        }
      }
    }
  }
}

// ---------------- scores: S[t,s] = SCALE * k_t . q_s, causal mask --------------
// q/k f16 [B,T,C] (heads interleaved). grid (T/64, T/64, B*H), block 128.
// Per-tile local accumulator + immediate store: no cross-tile phi copies.
__global__ __launch_bounds__(128) void attn_scores(
    const _Float16* __restrict__ kh, const _Float16* __restrict__ qh,
    float* __restrict__ S) {
  unsigned lane = threadIdx.x & 31;
  int wave = wave_id();
  int z = blockIdx.z, b = z / H_, h = z % H_;
  const _Float16* Kp = kh + (size_t)b * T_ * C_ + h * HD_;  // row t stride C_
  const _Float16* Qp = qh + (size_t)b * T_ * C_ + h * HD_;
  float* Sp = S + (size_t)z * T_ * T_;
  int row0 = (blockIdx.y * 4 + wave) * 16;   // t tile (scalar)
  int sb = blockIdx.x * 64;
  int hi = lane >> 4, nl = lane & 15;
  // Hoist A fragments (K rows) once: reused by all 4 s-tiles.
  v16h a0 = a_frag(Kp, C_, row0, 0, lane);
  v16h a1 = a_frag(Kp, C_, row0, 32, lane);
#pragma unroll
  for (int t = 0; t < 4; ++t) {
    int s0 = sb + 16 * t;
    int s = s0 + nl;
    v8f acc = {};
    if (s0 < row0 + 16) {                     // scalar branch: tile touches causal
      acc = wmma_f16(a0, b_frag(Qp, C_, s, 0, hi), acc);
      acc = wmma_f16(a1, b_frag(Qp, C_, s, 32, hi), acc);
    }
#pragma unroll
    for (int j = 0; j < 8; ++j) {
      int tt = row0 + j + 8 * hi;
      float v = (s <= tt) ? acc[j] * SCALE_ : -1e30f;
      Sp[(size_t)tt * T_ + s] = v;
    }
  }
}

// ---------------- row softmax over T=256 -> f16 attention ----------------------
__global__ __launch_bounds__(256) void softmax_rows(
    const float* __restrict__ S, _Float16* __restrict__ Ah) {
  unsigned lane = threadIdx.x & 31;
  size_t row = (size_t)blockIdx.x * 8 + wave_id();
  const float* p = S + row * T_;
  _Float16* o = Ah + row * T_;
  float v[8], mx = -1e30f;
#pragma unroll
  for (int i = 0; i < 8; ++i) { v[i] = p[lane + 32 * i]; mx = fmaxf(mx, v[i]); }
#pragma unroll
  for (int off = 16; off >= 1; off >>= 1) mx = fmaxf(mx, __shfl_xor(mx, off, 32));
  float s = 0.f;
#pragma unroll
  for (int i = 0; i < 8; ++i) { v[i] = __expf(v[i] - mx); s += v[i]; }
#pragma unroll
  for (int off = 16; off >= 1; off >>= 1) s += __shfl_xor(s, off, 32);
  float inv = 1.f / s;
#pragma unroll
  for (int i = 0; i < 8; ++i) o[lane + 32 * i] = (_Float16)(v[i] * inv);
}

// ---------------- ctx = att @ V, V pre-transposed f16 [B, C, T] ----------------
// grid (1, T/64, B*H); K = T = 256 -> 8 k-steps x 4 tiles = 32 WMMAs.
__global__ __launch_bounds__(128) void attn_av(
    const _Float16* __restrict__ Ah, const _Float16* __restrict__ vt,
    _Float16* __restrict__ ctx) {
  unsigned lane = threadIdx.x & 31;
  int wave = wave_id();
  int z = blockIdx.z, b = z / H_, h = z % H_;
  const _Float16* Ap = Ah + (size_t)z * T_ * T_;
  const _Float16* Vp = vt + ((size_t)b * C_ + h * HD_) * T_;  // row d stride T_
  int row0 = (blockIdx.y * 4 + wave) * 16;
  int hi = lane >> 4, nl = lane & 15;
  v8f acc[4] = {};
#pragma unroll
  for (int k0 = 0; k0 < T_; k0 += 32) {
    v16h a = a_frag(Ap, T_, row0, k0, lane);
#pragma unroll
    for (int t = 0; t < 4; ++t) {
      v16h bb = b_frag(Vp, T_, 16 * t + nl, k0, hi);
      acc[t] = wmma_f16(a, bb, acc[t]);
    }
  }
#pragma unroll
  for (int t = 0; t < 4; ++t) {
    int d = 16 * t + nl;
#pragma unroll
    for (int j = 0; j < 8; ++j) {
      int tt = row0 + j + 8 * hi;
      ctx[(((size_t)b * T_ + tt) * H_ + h) * HD_ + d] = (_Float16)acc[t][j];
    }
  }
}

// ---------------- x += LayerNorm(y); dual-store fp32 + f16 ---------------------
__global__ __launch_bounds__(256) void add_ln(
    float* __restrict__ x, _Float16* __restrict__ xh,
    const float* __restrict__ y,
    const float* __restrict__ g, const float* __restrict__ bta) {
  unsigned lane = threadIdx.x & 31;
  size_t row = (size_t)blockIdx.x * 8 + wave_id();
  const float* yr = y + row * C_;
  float* xr = x + row * C_;
  _Float16* hr = xh + row * C_;
  float v[12], s = 0.f;
#pragma unroll
  for (int i = 0; i < 12; ++i) { v[i] = yr[lane + 32 * i]; s += v[i]; }
#pragma unroll
  for (int o = 16; o >= 1; o >>= 1) s += __shfl_xor(s, o, 32);
  float mean = s * (1.f / C_);
  float var = 0.f;
#pragma unroll
  for (int i = 0; i < 12; ++i) { float d = v[i] - mean; var += d * d; }
#pragma unroll
  for (int o = 16; o >= 1; o >>= 1) var += __shfl_xor(var, o, 32);
  float inv = rsqrtf(var * (1.f / C_) + EPS_);
#pragma unroll
  for (int i = 0; i < 12; ++i) {
    int c = lane + 32 * i;
    float nv = xr[c] + (v[i] - mean) * inv * g[c] + bta[c];
    xr[c] = nv;
    hr[c] = (_Float16)nv;
  }
}

// ---------------- final LayerNorm -> f16 only ----------------------------------
__global__ __launch_bounds__(256) void ln_final(
    const float* __restrict__ x, _Float16* __restrict__ o,
    const float* __restrict__ g, const float* __restrict__ bta) {
  unsigned lane = threadIdx.x & 31;
  size_t row = (size_t)blockIdx.x * 8 + wave_id();
  const float* xr = x + row * C_;
  _Float16* orow = o + row * C_;
  float v[12], s = 0.f;
#pragma unroll
  for (int i = 0; i < 12; ++i) { v[i] = xr[lane + 32 * i]; s += v[i]; }
#pragma unroll
  for (int off = 16; off >= 1; off >>= 1) s += __shfl_xor(s, off, 32);
  float mean = s * (1.f / C_);
  float var = 0.f;
#pragma unroll
  for (int i = 0; i < 12; ++i) { float d = v[i] - mean; var += d * d; }
#pragma unroll
  for (int off = 16; off >= 1; off >>= 1) var += __shfl_xor(var, off, 32);
  float inv = rsqrtf(var * (1.f / C_) + EPS_);
#pragma unroll
  for (int i = 0; i < 12; ++i) {
    int c = lane + 32 * i;
    orow[c] = (_Float16)((v[i] - mean) * inv * g[c] + bta[c]);
  }
}

// ---------------- token embedding: fp32 + f16 ----------------------------------
__global__ void embed_kernel(const int* __restrict__ inp,
                             const float* __restrict__ tok,
                             const float* __restrict__ pos,
                             float* __restrict__ x, _Float16* __restrict__ xh) {
  size_t i = (size_t)blockIdx.x * blockDim.x + threadIdx.x;
  if (i >= (size_t)BT_ * C_) return;
  size_t row = i / C_;
  int c = (int)(i % C_);
  int t = (int)(row % T_);
  float v = tok[(size_t)inp[row] * C_ + c] + pos[(size_t)t * C_ + c];
  x[i] = v;
  xh[i] = (_Float16)v;
}

// ---------------- weight prepack: W[K,N] batches -> f16 Wt[Npad,K], zero-pad ---
__global__ void prep_wt(const float* __restrict__ W, _Float16* __restrict__ Wt,
                        int K, int N, int Npad, int nb) {
  size_t i = (size_t)blockIdx.x * blockDim.x + threadIdx.x;
  size_t per = (size_t)Npad * K;
  if (i >= per * nb) return;
  size_t bidx = i / per, rem = i % per;
  int n = (int)(rem / K), kk = (int)(rem % K);
  float v = (n < N) ? W[bidx * (size_t)K * N + (size_t)kk * N + n] : 0.f;
  Wt[i] = (_Float16)v;
}

// ---------------- per-row cross-entropy: rl[row] = lse - logit[target] ---------
__global__ __launch_bounds__(256) void row_loss(
    const float* __restrict__ logits, const int* __restrict__ tgt,
    float* __restrict__ rl) {
  unsigned lane = threadIdx.x & 31;
  size_t row = (size_t)blockIdx.x * 8 + wave_id();
  const float* p = logits + row * V_;
  float v[3], mx = -1e30f;
#pragma unroll
  for (int i = 0; i < 3; ++i) {
    int c = lane + 32 * i;
    v[i] = (c < V_) ? p[c] : -1e30f;
    mx = fmaxf(mx, v[i]);
  }
#pragma unroll
  for (int o = 16; o >= 1; o >>= 1) mx = fmaxf(mx, __shfl_xor(mx, o, 32));
  float s = 0.f;
#pragma unroll
  for (int i = 0; i < 3; ++i) {
    int c = lane + 32 * i;
    if (c < V_) s += __expf(v[i] - mx);
  }
#pragma unroll
  for (int o = 16; o >= 1; o >>= 1) s += __shfl_xor(s, o, 32);
  if (lane == 0) rl[row] = mx + __logf(s) - p[tgt[row]];
}

__global__ __launch_bounds__(256) void reduce_loss(const float* __restrict__ rl,
                                                   float* __restrict__ out) {
  __shared__ float sm[256];
  float s = 0.f;
  for (int i = threadIdx.x; i < BT_; i += 256) s += rl[i];
  sm[threadIdx.x] = s;
  __syncthreads();
  for (int o = 128; o >= 1; o >>= 1) {
    if ((int)threadIdx.x < o) sm[threadIdx.x] += sm[threadIdx.x + o];
    __syncthreads();
  }
  if (threadIdx.x == 0) out[0] = sm[0] * (1.f / BT_);
}

// ================================ host ========================================
extern "C" void kernel_launch(void* const* d_in, const int* in_sizes, int n_in,
                              void* d_out, int out_size, void* d_ws, size_t ws_size,
                              hipStream_t stream) {
  (void)in_sizes; (void)n_in; (void)out_size; (void)ws_size;
  const int*   inp  = (const int*)d_in[0];
  const int*   tgt  = (const int*)d_in[1];
  const float* tok  = (const float*)d_in[2];
  const float* pos  = (const float*)d_in[3];
  const float* Wq   = (const float*)d_in[4];
  const float* bq   = (const float*)d_in[5];
  const float* Wk   = (const float*)d_in[6];
  const float* bk   = (const float*)d_in[7];
  const float* Wv   = (const float*)d_in[8];
  const float* bv   = (const float*)d_in[9];
  const float* Wo   = (const float*)d_in[10];
  const float* bo   = (const float*)d_in[11];
  const float* ln1g = (const float*)d_in[12];
  const float* ln1b = (const float*)d_in[13];
  const float* W1   = (const float*)d_in[14];
  const float* b1   = (const float*)d_in[15];
  const float* W2   = (const float*)d_in[16];
  const float* b2   = (const float*)d_in[17];
  const float* ln2g = (const float*)d_in[18];
  const float* ln2b = (const float*)d_in[19];
  const float* lnfg = (const float*)d_in[20];
  const float* lnfb = (const float*)d_in[21];
  const float* Wlm  = (const float*)d_in[22];
  const float* blm  = (const float*)d_in[23];
  float* logits = (float*)d_out;                 // [BT, V]
  float* lossp  = logits + (size_t)BT_ * V_;

  // ---- workspace bump allocator (256B aligned) ----
  char* wp = (char*)d_ws;
  auto alloc = [&](size_t bytes) {
    void* p = (void*)wp;
    wp += (bytes + 255) & ~(size_t)255;
    return p;
  };
  float*     x    = (float*)alloc((size_t)BT_ * C_ * 4);          // residual fp32
  float*     S    = (float*)alloc((size_t)B_ * H_ * T_ * T_ * 4); // scores fp32
  float*     ao   = (float*)alloc((size_t)BT_ * C_ * 4);          // sublayer out
  float*     rl   = (float*)alloc((size_t)BT_ * 4);
  _Float16*  xh   = (_Float16*)alloc((size_t)BT_ * C_ * 2);       // f16 residual
  _Float16*  qh   = (_Float16*)alloc((size_t)BT_ * C_ * 2);       // [B,T,H*HD]
  _Float16*  kh   = (_Float16*)alloc((size_t)BT_ * C_ * 2);
  _Float16*  vt   = (_Float16*)alloc((size_t)BT_ * C_ * 2);       // [B, C, T]
  _Float16*  Ah   = (_Float16*)alloc((size_t)B_ * H_ * T_ * T_ * 2);
  _Float16*  ctxh = (_Float16*)alloc((size_t)BT_ * C_ * 2);       // [B,T,H,HD]
  _Float16*  f1h  = (_Float16*)alloc((size_t)BT_ * FF_ * 2);
  _Float16*  xfh  = (_Float16*)alloc((size_t)BT_ * C_ * 2);
  _Float16*  Wqt  = (_Float16*)alloc((size_t)L_ * C_ * C_ * 2);   // [L, H*HD, C]
  _Float16*  Wkt  = (_Float16*)alloc((size_t)L_ * C_ * C_ * 2);
  _Float16*  Wvt  = (_Float16*)alloc((size_t)L_ * C_ * C_ * 2);
  _Float16*  Wot  = (_Float16*)alloc((size_t)L_ * C_ * C_ * 2);   // [L, C, C]
  _Float16*  W1t  = (_Float16*)alloc((size_t)L_ * FF_ * C_ * 2);  // [L, FF, C]
  _Float16*  W2t  = (_Float16*)alloc((size_t)L_ * C_ * FF_ * 2);  // [L, C, FF]
  _Float16*  Wlmt = (_Float16*)alloc((size_t)VP_ * C_ * 2);       // [VP, C] padded

  dim3 blk128(128), blk256(256);
  auto cdiv = [](size_t a, size_t b) { return (unsigned)((a + b - 1) / b); };

  // ---- weight prepack (f16, transposed, padded) ----
  prep_wt<<<cdiv((size_t)L_ * H_ * HD_ * C_, 256), blk256, 0, stream>>>(Wq, Wqt, C_, HD_, HD_, L_ * H_);
  prep_wt<<<cdiv((size_t)L_ * H_ * HD_ * C_, 256), blk256, 0, stream>>>(Wk, Wkt, C_, HD_, HD_, L_ * H_);
  prep_wt<<<cdiv((size_t)L_ * H_ * HD_ * C_, 256), blk256, 0, stream>>>(Wv, Wvt, C_, HD_, HD_, L_ * H_);
  prep_wt<<<cdiv((size_t)L_ * C_ * C_, 256), blk256, 0, stream>>>(Wo, Wot, C_, C_, C_, L_);
  prep_wt<<<cdiv((size_t)L_ * FF_ * C_, 256), blk256, 0, stream>>>(W1, W1t, C_, FF_, FF_, L_);
  prep_wt<<<cdiv((size_t)L_ * C_ * FF_, 256), blk256, 0, stream>>>(W2, W2t, FF_, C_, C_, L_);
  prep_wt<<<cdiv((size_t)VP_ * C_, 256), blk256, 0, stream>>>(Wlm, Wlmt, C_, V_, VP_, 1);

  embed_kernel<<<cdiv((size_t)BT_ * C_, 256), blk256, 0, stream>>>(inp, tok, pos, x, xh);

  dim3 gC(C_ / 64, BT_ / 128);    // N=384 GEMMs, 128 rows per block
  dim3 gF(FF_ / 64, BT_ / 128);   // N=1536

  for (int l = 0; l < L_; ++l) {
    size_t wq = (size_t)l * C_ * C_, wf1 = (size_t)l * FF_ * C_;
    // Q, K row-major f16 [B,T,C]; V transposed f16 [B,C,T]
    gemm_k<1, true><<<gC, blk128, 0, stream>>>(xh, C_, Wqt + wq, bq + (size_t)l * C_,
                                               qh, C_, BT_, C_, C_, C_, 0);
    gemm_k<1, true><<<gC, blk128, 0, stream>>>(xh, C_, Wkt + wq, bk + (size_t)l * C_,
                                               kh, C_, BT_, C_, C_, C_, 0);
    gemm_k<2, true><<<gC, blk128, 0, stream>>>(xh, C_, Wvt + wq, bv + (size_t)l * C_,
                                               vt, C_, BT_, C_, C_, C_, 0);
    attn_scores<<<dim3(T_ / 64, T_ / 64, B_ * H_), blk128, 0, stream>>>(kh, qh, S);
    softmax_rows<<<(B_ * H_ * T_) / 8, blk256, 0, stream>>>(S, Ah);
    attn_av<<<dim3(1, T_ / 64, B_ * H_), blk128, 0, stream>>>(Ah, vt, ctxh);
    gemm_k<0, true><<<gC, blk128, 0, stream>>>(ctxh, C_, Wot + wq, bo + (size_t)l * C_,
                                               ao, C_, BT_, C_, C_, C_, 0);
    add_ln<<<BT_ / 8, blk256, 0, stream>>>(x, xh, ao, ln1g + (size_t)l * C_,
                                           ln1b + (size_t)l * C_);
    gemm_k<1, true><<<gF, blk128, 0, stream>>>(xh, C_, W1t + wf1, b1 + (size_t)l * FF_,
                                               f1h, FF_, BT_, FF_, FF_, C_, 1);
    gemm_k<0, true><<<gC, blk128, 0, stream>>>(f1h, FF_, W2t + wf1, b2 + (size_t)l * C_,
                                               ao, C_, BT_, C_, C_, FF_, 0);
    add_ln<<<BT_ / 8, blk256, 0, stream>>>(x, xh, ao, ln2g + (size_t)l * C_,
                                           ln2b + (size_t)l * C_);
  }

  ln_final<<<BT_ / 8, blk256, 0, stream>>>(x, xfh, lnfg, lnfb);
  // LM head: N=65 valid, Npad=80 (guarded variant)
  gemm_k<0, false><<<dim3(2, BT_ / 128), blk128, 0, stream>>>(xfh, C_, Wlmt, blm,
                                                              logits, V_, BT_, V_, VP_, C_, 0);

  row_loss<<<BT_ / 8, blk256, 0, stream>>>(logits, tgt, rl);
  reduce_loss<<<1, blk256, 0, stream>>>(rl, lossp);
}
